// SAModule_206158430239
// MI455X (gfx1250) — matrix operand order repaired and checked by reading, not compiled, not run on hardware
//
#include <hip/hip_runtime.h>
#include <hip/hip_bf16.h>

typedef __attribute__((ext_vector_type(16))) _Float16 v16h;
typedef __attribute__((ext_vector_type(8)))  float    v8f;

#define B_  4
#define C_  512
#define N_  4096
#define C8_ 64

union V16H { uint4 u4[2]; v16h h; };

__device__ __forceinline__ float redmax16(float v) {
  v = fmaxf(v, __shfl_xor(v, 1, 32));
  v = fmaxf(v, __shfl_xor(v, 2, 32));
  v = fmaxf(v, __shfl_xor(v, 4, 32));
  v = fmaxf(v, __shfl_xor(v, 8, 32));
  return v;
}
__device__ __forceinline__ float redsum16(float v) {
  v += __shfl_xor(v, 1, 32);
  v += __shfl_xor(v, 2, 32);
  v += __shfl_xor(v, 4, 32);
  v += __shfl_xor(v, 8, 32);
  return v;
}

// ---------------------------------------------------------------------------
// Kernel 1: fused q/k/v projection.
// Y[n, o] = sum_c X[b,c,n] * Wcat[o,c] + bcat[o],  o in [0,640):
//   o <  64 -> q  (row-major f16 [B,N,64])
//   o < 128 -> k  (row-major f16 [B,N,64])
//   else    -> v  (transposed f16 [B,C,N])
// One wave computes one 16x16 output tile; K-loop over C=512 in steps of 32.
// ---------------------------------------------------------------------------
__global__ void proj_kernel(const float* __restrict__ x,
                            const float* __restrict__ w1, const float* __restrict__ b1,
                            const float* __restrict__ w2, const float* __restrict__ b2,
                            const float* __restrict__ w3, const float* __restrict__ b3,
                            _Float16* __restrict__ qh, _Float16* __restrict__ kh,
                            _Float16* __restrict__ vh) {
  const int lane = threadIdx.x & 31;
  const int w    = threadIdx.x >> 5;
  const int h    = lane >> 4;        // half-wave
  const int m    = lane & 15;

  const int gw  = blockIdx.x * 4 + w;    // 4 * 256 * 40 = 40960 wave-tiles
  const int ot  = gw % 40;               // output-channel tile (16 wide)
  const int tmp = gw / 40;
  const int mt  = tmp % 256;             // spatial tile (16 rows)
  const int b   = tmp / 256;
  const int n0  = mt * 16;
  const int o0  = ot * 16;

  const float* wp; const float* bp; int osec;
  if (ot < 4)      { wp = w1; bp = b1; osec = o0;       }
  else if (ot < 8) { wp = w2; bp = b2; osec = o0 - 64;  }
  else             { wp = w3; bp = b3; osec = o0 - 128; }

  const float* xb = x + (size_t)b * C_ * (size_t)N_;
  v8f acc = {0.f, 0.f, 0.f, 0.f, 0.f, 0.f, 0.f, 0.f};

  for (int c0 = 0; c0 < C_; c0 += 32) {
    // A tile: X^T rows n0..n0+15, cols c0..c0+31 (16-bit A layout)
    v16h a;
#pragma unroll
    for (int v = 0; v < 4; ++v) {
      const int c = c0 + 16 * h + 2 * v;
      a[2 * v]     = (_Float16)xb[(size_t)c * N_ + n0 + m];
      a[2 * v + 1] = (_Float16)xb[(size_t)(c + 1) * N_ + n0 + m];
    }
#pragma unroll
    for (int v = 4; v < 8; ++v) {
      const int c = c0 + 16 * h + 2 * v;
      a[2 * v]     = (_Float16)xb[(size_t)c * N_ + n0 + m];
      a[2 * v + 1] = (_Float16)xb[(size_t)(c + 1) * N_ + n0 + m];
    }
    // B tile: W^T[c, o] = w[osec+m, c0+16h+2v(+1)] (16-bit B layout)
    v16h bm;
    const float2* wr2 = (const float2*)(wp + (size_t)(osec + m) * C_ + c0 + 16 * h);
#pragma unroll
    for (int v = 0; v < 8; ++v) {
      const float2 t = wr2[v];
      bm[2 * v]     = (_Float16)t.x;
      bm[2 * v + 1] = (_Float16)t.y;
    }
    acc = __builtin_amdgcn_wmma_f32_16x16x32_f16(false, a, false, bm,
                                                 (short)0, acc, false, false);
  }

  const float bias = bp[osec + m];
  if (ot < 8) {
    _Float16* dst = (ot < 4) ? qh : kh;
#pragma unroll
    for (int r = 0; r < 8; ++r) {
      const int row = n0 + r + 8 * h;
      dst[((size_t)b * N_ + row) * C8_ + osec + m] = (_Float16)(acc[r] + bias);
    }
  } else {
#pragma unroll
    for (int r = 0; r < 8; ++r) {
      const int row = n0 + r + 8 * h;
      vh[((size_t)(b * C_) + osec + m) * N_ + row] = (_Float16)(acc[r] + bias);
    }
  }
}

// ---------------------------------------------------------------------------
// Async stage of one 32-key x 64-dim f16 K block into LDS.
// 4 KB total = 256 x 16B chunks; 128 threads issue 2 async b128 loads each.
// Tracked by ASYNCcnt (CDNA5 async global->LDS path); loads complete in order.
// ---------------------------------------------------------------------------
__device__ __forceinline__ void stage_k_async(const _Float16* __restrict__ kh,
                                              const _Float16* kbuf_s,
                                              int b, int k0, int tid) {
#pragma unroll
  for (int i = 0; i < 2; ++i) {
    const int c = tid * 2 + i;                       // 16B chunk id, 0..255
    const unsigned lo = (unsigned)(uintptr_t)kbuf_s + 16u * (unsigned)c;
    const _Float16* gp =
        kh + ((size_t)b * N_ + k0 + (c >> 3)) * C8_ + (size_t)(c & 7) * 8;
    asm volatile("global_load_async_to_lds_b128 %0, %1, off"
                 :: "v"(lo), "v"(gp) : "memory");
  }
}

// ---------------------------------------------------------------------------
// Kernel 2: flash attention + residual.
// Block = 128 threads (4 waves) owns one 16-query tile; each wave owns a
// 128-wide d-slice (8 f32 accumulator tiles). K blocks are staged into LDS
// with double-buffered async DMA; online softmax over key blocks of 32.
// ---------------------------------------------------------------------------
__global__ void attn_kernel(const float* __restrict__ feat,
                            const _Float16* __restrict__ qh,
                            const _Float16* __restrict__ kh,
                            const _Float16* __restrict__ vh,
                            const float* __restrict__ gamma_p,
                            float* __restrict__ out) {
  __shared__ __align__(16) _Float16 pbuf[4][16 * 32];
  __shared__ __align__(16) _Float16 kbuf[2][32 * C8_];   // 2 x 4 KB, double buffer

  const int tid  = threadIdx.x;
  const int lane = tid & 31;
  const int w    = tid >> 5;
  const int h    = lane >> 4;
  const int m    = lane & 15;

  const int qt = blockIdx.x & 255;
  const int b  = blockIdx.x >> 8;
  const int n0 = qt * 16;
  const int dbase = w * 128;

  // Q tile: rows n0..n0+15, 64 head dims -> two f16 A tiles (contiguous 16B loads)
  const uint4* qv4 = (const uint4*)(qh + ((size_t)b * N_ + n0 + m) * C8_);
  V16H qa0, qa1;
  qa0.u4[0] = qv4[2 * h];     qa0.u4[1] = qv4[2 * h + 1];
  qa1.u4[0] = qv4[4 + 2 * h]; qa1.u4[1] = qv4[4 + 2 * h + 1];

  v8f O[8];
#pragma unroll
  for (int t = 0; t < 8; ++t) O[t] = (v8f){0.f,0.f,0.f,0.f,0.f,0.f,0.f,0.f};
  float mr[8], lr[8];
#pragma unroll
  for (int r = 0; r < 8; ++r) { mr[r] = -INFINITY; lr[r] = 0.f; }

  const v8f zf = {0.f,0.f,0.f,0.f,0.f,0.f,0.f,0.f};
  const int NKB = N_ / 32;

  // Prologue: DMA stage 0 into buffer 0.
  stage_k_async(kh, kbuf[0], b, 0, tid);

  for (int kb = 0; kb < NKB; ++kb) {
    const int k0 = kb * 32;

    // Issue DMA for the next K block into the other buffer, then wait until
    // only those 2 loads are outstanding -> current buffer is complete.
    if (kb + 1 < NKB) {
      stage_k_async(kh, kbuf[(kb + 1) & 1], b, (kb + 1) * 32, tid);
      asm volatile("s_wait_asynccnt 2" ::: "memory");
    } else {
      asm volatile("s_wait_asynccnt 0" ::: "memory");
    }
    __syncthreads();

    // K B-tiles from LDS: B[c, j] = kbuf[key j][c]; j0 in {0,16}, c0 in {0,32}
    const _Float16* kbb = kbuf[kb & 1];
    const uint4* kr0 = (const uint4*)(kbb + (size_t)m * C8_);
    const uint4* kr1 = (const uint4*)(kbb + (size_t)(16 + m) * C8_);
    V16H kb00, kb01, kb10, kb11;
    kb00.u4[0] = kr0[2 * h];     kb00.u4[1] = kr0[2 * h + 1];
    kb01.u4[0] = kr0[4 + 2 * h]; kb01.u4[1] = kr0[4 + 2 * h + 1];
    kb10.u4[0] = kr1[2 * h];     kb10.u4[1] = kr1[2 * h + 1];
    kb11.u4[0] = kr1[4 + 2 * h]; kb11.u4[1] = kr1[4 + 2 * h + 1];

    v8f s0 = __builtin_amdgcn_wmma_f32_16x16x32_f16(false, qa0.h, false, kb00.h,
                                                    (short)0, zf, false, false);
    s0     = __builtin_amdgcn_wmma_f32_16x16x32_f16(false, qa1.h, false, kb01.h,
                                                    (short)0, s0, false, false);
    v8f s1 = __builtin_amdgcn_wmma_f32_16x16x32_f16(false, qa0.h, false, kb10.h,
                                                    (short)0, zf, false, false);
    s1     = __builtin_amdgcn_wmma_f32_16x16x32_f16(false, qa1.h, false, kb11.h,
                                                    (short)0, s1, false, false);

    // Online softmax: row (r, half) lives across the 16 lanes of a half-wave.
#pragma unroll
    for (int r = 0; r < 8; ++r) {
      float t = fmaxf(s0[r], s1[r]);
      t = redmax16(t);
      const float mnew  = fmaxf(mr[r], t);
      const float alpha = __expf(mr[r] - mnew);
      mr[r] = mnew;
      const float p0 = __expf(s0[r] - mnew);
      const float p1 = __expf(s1[r] - mnew);
      lr[r] = lr[r] * alpha + redsum16(p0 + p1);
      pbuf[w][(r + 8 * h) * 32 + m]      = (_Float16)p0;
      pbuf[w][(r + 8 * h) * 32 + 16 + m] = (_Float16)p1;
#pragma unroll
      for (int t8 = 0; t8 < 8; ++t8) O[t8][r] *= alpha;
    }
    __syncthreads();

    // Re-read P in f16 A-layout (row m, two 16B chunks)
    V16H pa;
    const uint4* pv = (const uint4*)&pbuf[w][m * 32];
    pa.u4[0] = pv[2 * h];
    pa.u4[1] = pv[2 * h + 1];

    // O[:, d-slice] += P @ V ; vh is [C, N] so key dim is contiguous
#pragma unroll
    for (int t = 0; t < 8; ++t) {
      const int d = dbase + t * 16 + m;
      const uint4* vr = (const uint4*)(vh + ((size_t)(b * C_) + d) * N_ + k0);
      V16H vb;
      vb.u4[0] = vr[2 * h];
      vb.u4[1] = vr[2 * h + 1];
      O[t] = __builtin_amdgcn_wmma_f32_16x16x32_f16(false, pa.h, false, vb.h,
                                                    (short)0, O[t], false, false);
    }
    __syncthreads();   // protect pbuf and current kbuf buffer before reuse
  }

  const float g = gamma_p[0];
  float invl[8];
#pragma unroll
  for (int r = 0; r < 8; ++r) invl[r] = 1.f / lr[r];

#pragma unroll
  for (int t = 0; t < 8; ++t) {
    const int d = dbase + t * 16 + m;
#pragma unroll
    for (int r = 0; r < 8; ++r) {
      const int n = n0 + r + 8 * h;
      const size_t idx = ((size_t)(b * C_) + d) * N_ + n;
      out[idx] = g * (O[t][r] * invl[r]) + feat[idx];
    }
  }
}

extern "C" void kernel_launch(void* const* d_in, const int* in_sizes, int n_in,
                              void* d_out, int out_size, void* d_ws, size_t ws_size,
                              hipStream_t stream) {
  const float* feat  = (const float*)d_in[0];
  const float* w1    = (const float*)d_in[1];
  const float* b1    = (const float*)d_in[2];
  const float* w2    = (const float*)d_in[3];
  const float* b2    = (const float*)d_in[4];
  const float* w3    = (const float*)d_in[5];
  const float* b3    = (const float*)d_in[6];
  const float* gamma = (const float*)d_in[7];
  float* out = (float*)d_out;

  char* ws = (char*)d_ws;
  _Float16* qh = (_Float16*)(ws);                       //  2 MB
  _Float16* kh = (_Float16*)(ws + (size_t)(2  << 20));  //  2 MB
  _Float16* vh = (_Float16*)(ws + (size_t)(4  << 20));  // 16 MB

  // 4 batches * 256 n-tiles * 40 o-tiles = 40960 wave-tiles, 4 waves/block
  proj_kernel<<<dim3(10240), dim3(128), 0, stream>>>(feat, w1, b1, w2, b2, w3, b3,
                                                     qh, kh, vh);
  // 4 batches * 256 query tiles, 4 waves/block (d split 4 x 128)
  attn_kernel<<<dim3(1024), dim3(128), 0, stream>>>(feat, qh, kh, vh, gamma, out);
}
